// MultiheadAttention_50998441673227
// MI455X (gfx1250) — matrix operand order repaired and checked by reading, not compiled, not run on hardware
//
#include <hip/hip_runtime.h>
#include <hip/hip_bf16.h>
#include <math.h>

// MI455X MHA forward. ~88 MiB workspace:
//   xb(16M) | WqkvT_b(6M) | WoT_b(2M) | qkv_b(48M) | vals_b(16M)
// All matmuls on v_wmma_f32_16x16x32_bf16 (fp32 accumulate). Flash-attention
// streaming softmax. All LDS tiles are contraction-major so every WMMA
// fragment read is ds_load_b128; softmax row reductions use v_permlane16.

typedef __bf16 v16bf  __attribute__((ext_vector_type(16)));
typedef __bf16 bf16x8 __attribute__((ext_vector_type(8)));
typedef float  v8f    __attribute__((ext_vector_type(8)));

#define D_MODEL 1024
#define NUM_HEAD 16
#define HEAD_DIM 64
#define BATCH 4
#define SEQ 2048
#define QKV_N (3 * D_MODEL)

// --- WMMA fragment index helpers (wave32, ISA 7.12.2) ---
// A 16x32 bf16: lane l holds row M=l%16; element e -> K index (two
// contiguous 8-runs per lane: {0..7,16..23} lanes 0-15, {8..15,24..31} hi).
__device__ __forceinline__ int a_k_of(int lane, int e) {
  int lh = (lane >> 4) & 1;
  int vg = e >> 1;
  return ((vg & 4) << 2) + lh * 8 + ((vg & 3) << 1) + (e & 1);
}
// B 32x16: lane l holds col N=l%16; same K order as A: {0..7,16..23} lanes 0-15, {8..15,24..31} hi
__device__ __forceinline__ int b_k_of(int lane, int e) {
  return ((e >> 3) << 4) + (((lane >> 4) & 1) << 3) + (e & 7);
}
// C/D 16x16 f32: vgpr r, lane l -> row = r + 8*(l>=16), col = l%16

// xor-butterfly within each 16-lane half, on the VALU (v_permlane16_b32)
// instead of the DS pipe (ds_bpermute).  Lane-select nibbles: sel(i)=i^M.
template <int M>
__device__ __forceinline__ float lane_xor16(float x) {
#if __has_builtin(__builtin_amdgcn_permlane16)
  constexpr unsigned lo = (M == 1) ? 0x67452301u
                        : (M == 2) ? 0x54761032u
                        : (M == 4) ? 0x32107654u
                                   : 0xFEDCBA98u;
  constexpr unsigned hi = (M == 1) ? 0xEFCDAB89u
                        : (M == 2) ? 0xDCFE98BAu
                        : (M == 4) ? 0xBA98FEDCu
                                   : 0x76543210u;
  unsigned u = __float_as_uint(x);
  u = __builtin_amdgcn_permlane16(u, u, lo, hi, false, false);
  return __uint_as_float(u);
#else
  return __shfl_xor(x, M, 32);
#endif
}
__device__ __forceinline__ float half_reduce_max(float t) {
  t = fmaxf(t, lane_xor16<1>(t));
  t = fmaxf(t, lane_xor16<2>(t));
  t = fmaxf(t, lane_xor16<4>(t));
  t = fmaxf(t, lane_xor16<8>(t));
  return t;
}
__device__ __forceinline__ float half_reduce_sum(float t) {
  t += lane_xor16<1>(t);
  t += lane_xor16<2>(t);
  t += lane_xor16<4>(t);
  t += lane_xor16<8>(t);
  return t;
}

__global__ void cvt_f32_bf16(const float* __restrict__ src,
                             __bf16* __restrict__ dst, int n) {
  int i = blockIdx.x * blockDim.x + threadIdx.x;
  if (i < n) dst[i] = (__bf16)src[i];
}

// src[K=1024][N] fp32  ->  dst[N][K=1024] bf16 (pre-transposed weights so the
// GEMM can stage the B tile contraction-major with b128 on both sides).
__global__ void cvt_f32_bf16_T(const float* __restrict__ src,
                               __bf16* __restrict__ dst, int total, int N) {
  int i = blockIdx.x * blockDim.x + threadIdx.x;
  if (i < total) {
    int n = i >> 10;        // K = 1024
    int k = i & 1023;
    dst[i] = (__bf16)src[(size_t)k * N + n];
  }
}

// C[M,N] = A[M,K] * Bt[N,K]^T + bias[N].  Block 256 thr, tile 64(M) x 128(N).
// Wave w: rows (w%4)*16, cols (w/4)*64 -> 4 WMMA C tiles. Double-buffered
// staging; both LDS tiles contraction-major.
template <bool OUT_BF16>
__global__ __launch_bounds__(256) void gemm_bf16_wmma(
    const __bf16* __restrict__ A, const __bf16* __restrict__ Bt,
    const float* __restrict__ bias, void* __restrict__ Cp,
    int M, int N, int K) {
  __shared__ __bf16 As[64][40];    // [m][k] tile, padded
  __shared__ __bf16 Bs[128][40];   // [n][k] tile, padded (contraction-major)
  const int tid  = threadIdx.x;
  const int lane = tid & 31;
  const int wave = tid >> 5;
  const int lh   = (lane >> 4) & 1;
  const int c16  = lane & 15;
  const int m0 = blockIdx.y * 64;
  const int n0 = blockIdx.x * 128;
  const int wm = (wave & 3) * 16;
  const int wn = (wave >> 2) * 64;
  const int ar = tid >> 2, ac = (tid & 3) * 8;    // A staging: 64 m x 32 k
  const int bn = tid >> 1, bk = (tid & 1) * 16;   // B staging: 128 n x 32 k

  v8f acc[4] = {};
  bf16x8 areg, breg0, breg1;
  areg  = *(const bf16x8*)&A[(size_t)(m0 + ar) * K + ac];
  breg0 = *(const bf16x8*)&Bt[(size_t)(n0 + bn) * K + bk];
  breg1 = *(const bf16x8*)&Bt[(size_t)(n0 + bn) * K + bk + 8];

  for (int k0 = 0; k0 < K; k0 += 32) {
    *(bf16x8*)&As[ar][ac]     = areg;
    *(bf16x8*)&Bs[bn][bk]     = breg0;
    *(bf16x8*)&Bs[bn][bk + 8] = breg1;
    __syncthreads();
    if (k0 + 32 < K) {   // prefetch next tile while this one computes
      areg  = *(const bf16x8*)&A[(size_t)(m0 + ar) * K + k0 + 32 + ac];
      breg0 = *(const bf16x8*)&Bt[(size_t)(n0 + bn) * K + k0 + 32 + bk];
      breg1 = *(const bf16x8*)&Bt[(size_t)(n0 + bn) * K + k0 + 32 + bk + 8];
    }
    v16bf af;
#pragma unroll
    for (int e = 0; e < 16; ++e) af[e] = As[wm + c16][a_k_of(lane, e)];
#pragma unroll
    for (int nt = 0; nt < 4; ++nt) {
      v16bf bfr;
#pragma unroll
      for (int e = 0; e < 16; ++e) bfr[e] = Bs[wn + nt * 16 + c16][b_k_of(lane, e)];
      acc[nt] = __builtin_amdgcn_wmma_f32_16x16x32_bf16(
          false, af, false, bfr, (short)0, acc[nt], false, false);
    }
    __syncthreads();
  }
#pragma unroll
  for (int nt = 0; nt < 4; ++nt) {
    int col = n0 + wn + nt * 16 + c16;
    float bv = bias[col];
#pragma unroll
    for (int r = 0; r < 8; ++r) {
      int row = m0 + wm + lh * 8 + r;
      float v = acc[nt][r] + bv;
      if (OUT_BF16) ((__bf16*)Cp)[(size_t)row * N + col] = (__bf16)v;
      else          ((float*)Cp)[(size_t)row * N + col] = v;
    }
  }
}

// Flash attention: grid = B*H*(S/128); block 256 = 8 waves; wave owns 16 q-rows.
// qkv row layout: head h -> [h*192 .. h*192+191] = [Q(64) K(64) V(64)].
__global__ __launch_bounds__(256) void attn_fwd(
    const __bf16* __restrict__ qkv, const float* __restrict__ mask,
    __bf16* __restrict__ vals) {
  __shared__ __bf16 Kl[32][72];      // [key][d]   (contraction d for Q*K^T)
  __shared__ __bf16 Vt[64][40];      // [d][key]   (contraction key for P*V)
  __shared__ __bf16 Pl[8][16][32];   // per-wave P tile for C->A relayout

  const int tid  = threadIdx.x;
  const int lane = tid & 31;
  const int wave = tid >> 5;
  const int lh   = (lane >> 4) & 1;
  const int c16  = lane & 15;
  const int qblk = blockIdx.x & 15;
  const int h    = (blockIdx.x >> 4) & 15;
  const int b    = blockIdx.x >> 8;
  const int sq0  = qblk * 128 + wave * 16;
  const float scale = 0.125f;  // 1/sqrt(64)
  const size_t brow = (size_t)b * SEQ;

  // Q A-fragments (two K-chunks of 32 over d=64) straight from global.
  v16bf qf[2];
  {
    const __bf16* qb = qkv + (brow + sq0 + c16) * QKV_N + h * 3 * HEAD_DIM;
#pragma unroll
    for (int kc = 0; kc < 2; ++kc)
#pragma unroll
      for (int e = 0; e < 16; ++e)
        qf[kc][e] = qb[kc * 32 + a_k_of(lane, e)];
  }

  v8f acc[4] = {};           // O accumulator: 16 x 64 as 4 C tiles
  float mrun[8], lrun[8];
#pragma unroll
  for (int r = 0; r < 8; ++r) { mrun[r] = -__builtin_inff(); lrun[r] = 0.f; }

  const int str = tid >> 3;        // staging: 32 keys x (8 thr x 8 d)
  const int stc = (tid & 7) * 8;

  bf16x8 kreg, vreg;
  {
    const __bf16* kvb = qkv + (brow + str) * QKV_N + h * 3 * HEAD_DIM;
    kreg = *(const bf16x8*)&kvb[HEAD_DIM + stc];
    vreg = *(const bf16x8*)&kvb[2 * HEAD_DIM + stc];
  }

  for (int sk0 = 0; sk0 < SEQ; sk0 += 32) {
    // commit the prefetched tile to LDS (V scattered contraction-major)
    *(bf16x8*)&Kl[str][stc] = kreg;
#pragma unroll
    for (int j = 0; j < 8; ++j) Vt[stc + j][str] = vreg[j];
    __syncthreads();

    if (sk0 + 32 < SEQ) {   // next tile's global loads overlap the compute
      const __bf16* kvb = qkv + (brow + sk0 + 32 + str) * QKV_N + h * 3 * HEAD_DIM;
      kreg = *(const bf16x8*)&kvb[HEAD_DIM + stc];
      vreg = *(const bf16x8*)&kvb[2 * HEAD_DIM + stc];
    }
    if (sk0 + 64 < SEQ) {   // L2 warmup two tiles ahead -> global_prefetch_b8
      __builtin_prefetch(qkv + (brow + sk0 + 64 + str) * QKV_N +
                         h * 3 * HEAD_DIM + HEAD_DIM + stc, 0, 1);
    }

    // S = Q * K^T : two 16-key N-tiles, contraction d=64 in two chunks.
    v8f c0 = {}, c1 = {};
#pragma unroll
    for (int kc = 0; kc < 2; ++kc) {
      v16bf kb0, kb1;
#pragma unroll
      for (int e = 0; e < 16; ++e) {
        int d = kc * 32 + b_k_of(lane, e);
        kb0[e] = Kl[c16][d];
        kb1[e] = Kl[16 + c16][d];
      }
      c0 = __builtin_amdgcn_wmma_f32_16x16x32_bf16(false, qf[kc], false, kb0,
                                                   (short)0, c0, false, false);
      c1 = __builtin_amdgcn_wmma_f32_16x16x32_bf16(false, qf[kc], false, kb1,
                                                   (short)0, c1, false, false);
    }

    // scale + mask + online softmax (row M spans the 16 lanes of a half)
#pragma unroll
    for (int r = 0; r < 8; ++r) {
      const size_t mr = (size_t)(sq0 + lh * 8 + r) * SEQ + sk0;
      float s0 = c0[r] * scale + mask[mr + c16];
      float s1 = c1[r] * scale + mask[mr + 16 + c16];
      float mn = fmaxf(mrun[r], half_reduce_max(fmaxf(s0, s1)));
      float alpha = __expf(mrun[r] - mn);
      float p0 = __expf(s0 - mn);
      float p1 = __expf(s1 - mn);
      float rs = half_reduce_sum(p0 + p1);
      lrun[r] = lrun[r] * alpha + rs;
      mrun[r] = mn;
#pragma unroll
      for (int dt = 0; dt < 4; ++dt) acc[dt][r] = acc[dt][r] * alpha;
      Pl[wave][lh * 8 + r][c16]      = (__bf16)p0;
      Pl[wave][lh * 8 + r][16 + c16] = (__bf16)p1;
    }
    // wave-local LDS RAW: P written above is re-read as an A-fragment
    asm volatile("s_wait_dscnt 0" ::: "memory");

    v16bf pf;
#pragma unroll
    for (int e = 0; e < 16; ++e) pf[e] = Pl[wave][c16][a_k_of(lane, e)];
#pragma unroll
    for (int dt = 0; dt < 4; ++dt) {
      v16bf vb;
#pragma unroll
      for (int e = 0; e < 16; ++e) vb[e] = Vt[dt * 16 + c16][b_k_of(lane, e)];
      acc[dt] = __builtin_amdgcn_wmma_f32_16x16x32_bf16(false, pf, false, vb,
                                                        (short)0, acc[dt], false, false);
    }
    __syncthreads();
  }

  // O /= l ; store bf16 to vals[b, s, h*64 + d]
#pragma unroll
  for (int dt = 0; dt < 4; ++dt) {
    int col = h * HEAD_DIM + dt * 16 + c16;
#pragma unroll
    for (int r = 0; r < 8; ++r) {
      int row = sq0 + lh * 8 + r;
      float v = acc[dt][r] / lrun[r];
      vals[(brow + row) * (size_t)D_MODEL + col] = (__bf16)v;
    }
  }
}

extern "C" void kernel_launch(void* const* d_in, const int* in_sizes, int n_in,
                              void* d_out, int out_size, void* d_ws, size_t ws_size,
                              hipStream_t stream) {
  (void)in_sizes; (void)n_in; (void)out_size; (void)ws_size;
  const float* x    = (const float*)d_in[0];
  const float* mask = (const float*)d_in[1];
  const float* Wqkv = (const float*)d_in[2];
  const float* bqkv = (const float*)d_in[3];
  const float* Wo   = (const float*)d_in[4];
  const float* bo   = (const float*)d_in[5];
  float* out = (float*)d_out;

  const size_t n_x    = (size_t)BATCH * SEQ * D_MODEL;  // 8Mi
  const size_t n_wqkv = (size_t)D_MODEL * QKV_N;        // 3Mi
  const size_t n_wo   = (size_t)D_MODEL * D_MODEL;      // 1Mi
  const size_t n_qkv  = (size_t)BATCH * SEQ * QKV_N;    // 24Mi
  char* ws = (char*)d_ws;
  __bf16* xb     = (__bf16*)ws;  ws += n_x * 2;
  __bf16* wqkvtb = (__bf16*)ws;  ws += n_wqkv * 2;   // [3072][1024] (transposed)
  __bf16* wotb   = (__bf16*)ws;  ws += n_wo * 2;     // [1024][1024] (transposed)
  __bf16* qkvb   = (__bf16*)ws;  ws += n_qkv * 2;
  __bf16* valsb  = (__bf16*)ws;

  cvt_f32_bf16<<<(unsigned)((n_x + 255) / 256), 256, 0, stream>>>(x, xb, (int)n_x);
  cvt_f32_bf16_T<<<(unsigned)((n_wqkv + 255) / 256), 256, 0, stream>>>(
      Wqkv, wqkvtb, (int)n_wqkv, QKV_N);
  cvt_f32_bf16_T<<<(unsigned)((n_wo + 255) / 256), 256, 0, stream>>>(
      Wo, wotb, (int)n_wo, D_MODEL);

  // QKV projection: [8192,1024] x [1024,3072] + bqkv -> bf16
  gemm_bf16_wmma<true><<<dim3(QKV_N / 128, (BATCH * SEQ) / 64), 256, 0, stream>>>(
      xb, wqkvtb, bqkv, (void*)qkvb, BATCH * SEQ, QKV_N, D_MODEL);

  // Flash attention -> vals bf16 [8192,1024]
  attn_fwd<<<dim3(BATCH * NUM_HEAD * (SEQ / 128)), 256, 0, stream>>>(qkvb, mask, valsb);

  // Output projection: [8192,1024] x [1024,1024] + bo -> fp32 out
  gemm_bf16_wmma<false><<<dim3(D_MODEL / 128, (BATCH * SEQ) / 64), 256, 0, stream>>>(
      valsb, wotb, bo, (void*)out, BATCH * SEQ, D_MODEL, D_MODEL);
}
